// SparseTransformerLayer_56160992362952
// MI455X (gfx1250) — matrix operand order repaired and checked by reading, not compiled, not run on hardware
//
#include <hip/hip_runtime.h>

// Problem constants (from reference)
#define BB 4
#define SS 1024
#define DD 512
#define FF 2048
#define HH 8
#define EE 8
#define HDIM 64
#define CAP 256
#define NT (BB*SS)          // 4096 tokens

typedef __bf16 v16bf __attribute__((ext_vector_type(16)));
typedef float  v8f   __attribute__((ext_vector_type(8)));

union BF16x16 { v16bf v; unsigned u[8]; };

__device__ __forceinline__ unsigned short f2bf(float f) {
  unsigned u = __float_as_uint(f);
  u += 0x7FFFu + ((u >> 16) & 1u);       // round-to-nearest-even
  return (unsigned short)(u >> 16);
}
__device__ __forceinline__ unsigned pack2(float lo, float hi) {
  return (unsigned)f2bf(lo) | ((unsigned)f2bf(hi) << 16);
}

// CDNA5 async global->LDS copy, 16 bytes per lane (ASYNCcnt-tracked).
// Generic LDS pointer low 32 bits == LDS byte offset (ISA 10.2 aperture rule).
__device__ __forceinline__ void async_copy_b128(void* lds_dst, const void* gsrc) {
  unsigned ldst = (unsigned)(size_t)lds_dst;
  unsigned long long ga = (unsigned long long)gsrc;
  asm volatile("global_load_async_to_lds_b128 %0, %1, off" :: "v"(ldst), "v"(ga) : "memory");
}
__device__ __forceinline__ void wait_async0() {
  asm volatile("s_wait_asynccnt 0x0" ::: "memory");
}

// A-fragment: 16x32 bf16, M x K.  lanes 0-15: row=lane, K {0..7,16..23};
// lanes 16-31: row=lane-16, K {8..15,24..31}. Pairs (k,k+1) are one dword.
__device__ __forceinline__ void load_afrag(const unsigned short* base, int stride, unsigned* u) {
  int lane = threadIdx.x & 31;
  int row  = lane & 15;
  int koff = (lane < 16) ? 0 : 8;
  #pragma unroll
  for (int i = 0; i < 8; ++i) {
    int k = ((i < 4) ? (2 * i) : (16 + 2 * (i - 4))) + koff;
    u[i] = *(const unsigned*)(base + row * stride + k);
  }
}
// B-fragment: 32x16 bf16, K x N, stored in LDS as [N][K] (K contiguous).
// lanes 0-15: N=lane, K {0..7,16..23} ; lanes 16-31: N=lane-16, K {8..15,24..31}.
__device__ __forceinline__ void load_bfrag(const unsigned short* base, int stride, unsigned* u) {
  int lane = threadIdx.x & 31;
  int n  = lane & 15;
  int kb = (lane < 16) ? 0 : 8;
  #pragma unroll
  for (int i = 0; i < 8; ++i)
    u[i] = *(const unsigned*)(base + n * stride + kb + ((i < 4) ? (2 * i) : (16 + 2 * (i - 4))));
}

__device__ __forceinline__ v8f wmma_bf16(const BF16x16& a, const BF16x16& b, v8f c) {
  return __builtin_amdgcn_wmma_f32_16x16x32_bf16(false, a.v, false, b.v, (short)0, c, false, false);
}

// ---------------------------------------------------------------------------
// One-shot weight conversion f32 -> bf16 (plain, weights already [N][K])
// ---------------------------------------------------------------------------
__global__ __launch_bounds__(256) void cvt_bf16_kernel(
    const float* __restrict__ src, unsigned short* __restrict__ dst, long long n)
{
  long long i = (long long)blockIdx.x * 256 + threadIdx.x;
  if (i < n) dst[i] = f2bf(src[i]);
}
// Transpose-convert: src [z][Kd][Nd] f32 -> dst [z][Nd][Kd] bf16
__global__ __launch_bounds__(256) void cvt_bf16_t_kernel(
    const float* __restrict__ src, unsigned short* __restrict__ dst, int Kd, int Nd)
{
  long long i = (long long)blockIdx.x * 256 + threadIdx.x;   // over Nd*Kd
  int n = (int)(i / Kd), k = (int)(i % Kd);
  const float* s = src + (size_t)blockIdx.z * Kd * Nd;
  unsigned short* d = dst + (size_t)blockIdx.z * Kd * Nd;
  d[(size_t)n * Kd + k] = f2bf(s[(size_t)k * Nd + n]);
}

// ---------------------------------------------------------------------------
// bf16-WMMA GEMM:  C[M,N] = act( A[M,K] @ B[N,K]^T + bias )
// Block tile 128x64, K-step 32, 8 waves; each wave computes a 32x32 output
// (2x2 register blocking -> 4 WMMAs / K-step with 2x fragment reuse).
// B is always bf16 [N][K] and staged with GLOBAL_LOAD_ASYNC_TO_LDS_B128.
// A is f32 (converted in-stage) or bf16 (async-staged); C is f32 or bf16.
// ---------------------------------------------------------------------------
__global__ __launch_bounds__(256) void gemm_bf16_kernel(
    const void* __restrict__ Ain, const unsigned short* __restrict__ Bm,
    const float* __restrict__ bias, void* __restrict__ Cout,
    int K, int lda, int ldb, int ldc, int act, int a_bf16, int c_bf16,
    long long sA, long long sB, long long sC)
{
  __shared__ __align__(16) unsigned short As[128][32];   // 8 KB
  __shared__ __align__(16) unsigned short Bs[64][32];    // 4 KB, [N][K]
  const int tid  = threadIdx.x;
  const int lane = tid & 31;
  const int wave = tid >> 5;
  const int mi = wave & 3, ni = wave >> 2;       // 4x2 waves of 32x32 tiles
  const int m0 = blockIdx.y * 128, n0 = blockIdx.x * 64;
  const float* Af = nullptr;
  const unsigned short* A16 = nullptr;
  if (a_bf16) A16 = (const unsigned short*)Ain + (size_t)blockIdx.z * sA;
  else        Af  = (const float*)Ain + (size_t)blockIdx.z * sA;
  Bm += (size_t)blockIdx.z * sB;

  v8f acc[2][2];
  acc[0][0] = {}; acc[0][1] = {}; acc[1][0] = {}; acc[1][1] = {};

  for (int k0 = 0; k0 < K; k0 += 32) {
    __syncthreads();
    // B stage: 64x32 bf16 = 4 KB, one 16-byte async copy per thread
    {
      int n = tid >> 2, q = tid & 3;
      async_copy_b128(&Bs[n][q * 8], Bm + (size_t)(n0 + n) * ldb + k0 + q * 8);
    }
    if (a_bf16) {
      // A stage: 128x32 bf16 = 8 KB, two async copies per thread
      #pragma unroll
      for (int j = 0; j < 2; ++j) {
        int slot = tid * 2 + j;
        int m = slot >> 2, q = slot & 3;
        async_copy_b128(&As[m][q * 8], A16 + (size_t)(m0 + m) * lda + k0 + q * 8);
      }
    } else {
      // A stage: f32 -> bf16 conversion
      #pragma unroll
      for (int j = 0; j < 16; ++j) {
        int e = tid * 16 + j;
        int m = e >> 5, k = e & 31;
        As[m][k] = f2bf(Af[(size_t)(m0 + m) * lda + k0 + k]);
      }
      if (k0 + 32 < K)   // speculative prefetch of next A tile -> global_prefetch_b8
        __builtin_prefetch((const void*)(Af + (size_t)(m0 + ((tid * 16) >> 5)) * lda + k0 + 32), 0, 1);
    }
    wait_async0();
    __syncthreads();

    BF16x16 af0, af1, bf0, bf1;
    load_afrag(&As[mi * 32][0],      32, af0.u);
    load_afrag(&As[mi * 32 + 16][0], 32, af1.u);
    load_bfrag(&Bs[ni * 32][0],      32, bf0.u);
    load_bfrag(&Bs[ni * 32 + 16][0], 32, bf1.u);
    acc[0][0] = wmma_bf16(af0, bf0, acc[0][0]);
    acc[0][1] = wmma_bf16(af0, bf1, acc[0][1]);
    acc[1][0] = wmma_bf16(af1, bf0, acc[1][0]);
    acc[1][1] = wmma_bf16(af1, bf1, acc[1][1]);
  }

  const int col = lane & 15;
  const int hl8 = (lane < 16) ? 0 : 8;
  #pragma unroll
  for (int a = 0; a < 2; ++a) {
    int rbase = m0 + mi * 32 + a * 16 + hl8;
    #pragma unroll
    for (int bn = 0; bn < 2; ++bn) {
      int gcol = n0 + ni * 32 + bn * 16 + col;
      float bv = bias ? bias[gcol] : 0.0f;
      #pragma unroll
      for (int i = 0; i < 8; ++i) {
        float v = acc[a][bn][i] + bv;
        if (act) v = 0.5f * v * (1.0f + erff(v * 0.70710678118f));  // exact GELU
        size_t ci = (size_t)(rbase + i) * ldc + gcol;
        if (c_bf16) ((unsigned short*)Cout)[(size_t)blockIdx.z * sC + ci] = f2bf(v);
        else        ((float*)Cout)[(size_t)blockIdx.z * sC + ci] = v;
      }
    }
  }
}

// ---------------------------------------------------------------------------
// Flash attention: grid (S/64, B*H), 128 threads = 4 waves.
// Each wave owns a 16-query tile; block shares 32-key K/V chunks in LDS.
// ---------------------------------------------------------------------------
__global__ __launch_bounds__(128) void attn_kernel(
    const float* __restrict__ qkv, float* __restrict__ attn_out)
{
  const int D3 = 3 * DD;
  const int b = blockIdx.y >> 3;
  const int h = blockIdx.y & 7;
  const int wave = threadIdx.x >> 5;
  const int lane = threadIdx.x & 31;
  const int hl  = (lane < 16) ? 0 : 1;
  const int col = lane & 15;

  __shared__ unsigned short Kt[32][HDIM];       // [key][ch]  -> B frag for Q.K^T
  __shared__ unsigned short Vt[HDIM][32];       // [ch][key]  -> B frag for P.V
  __shared__ unsigned short Pl[4][16][32];      // per-wave P bounce

  // Q A-fragments, two 32-channel chunks, 1/sqrt(hd) folded in
  unsigned qa[2][8];
  {
    int qrow = blockIdx.x * 64 + wave * 16 + col;
    const float* qb = qkv + (size_t)(b * SS + qrow) * D3 + h * HDIM;
    int koff = hl ? 8 : 0;
    #pragma unroll
    for (int c = 0; c < 2; ++c)
      #pragma unroll
      for (int i = 0; i < 8; ++i) {
        int k = ((i < 4) ? (2 * i) : (16 + 2 * (i - 4))) + koff + 32 * c;
        qa[c][i] = pack2(qb[k] * 0.125f, qb[k + 1] * 0.125f);
      }
  }

  float m_run[8], l_run[8];
  v8f o[4]; o[0] = {}; o[1] = {}; o[2] = {}; o[3] = {};
  #pragma unroll
  for (int i = 0; i < 8; ++i) { m_run[i] = -1e30f; l_run[i] = 0.0f; }

  for (int kc = 0; kc < SS / 32; ++kc) {
    __syncthreads();
    const int kk0 = kc * 32;
    #pragma unroll
    for (int j = 0; j < 16; ++j) {              // 32x64 K and V tiles
      int e2 = threadIdx.x * 16 + j;
      int key = e2 >> 6, ch = e2 & 63;
      size_t base = (size_t)(b * SS + kk0 + key) * D3 + h * HDIM + ch;
      Kt[key][ch] = f2bf(qkv[base + DD]);
      Vt[ch][key] = f2bf(qkv[base + 2 * DD]);
    }
    __syncthreads();

    // scores: 16 queries x 32 keys (two 16x16 tiles), K-dim = 64 channels
    v8f s0 = {}, s1 = {};
    #pragma unroll
    for (int cc = 0; cc < 2; ++cc) {
      BF16x16 aq;
      #pragma unroll
      for (int i = 0; i < 8; ++i) aq.u[i] = qa[cc][i];
      BF16x16 b0, b1;
      load_bfrag(&Kt[0][cc * 32], HDIM, b0.u);
      load_bfrag(&Kt[16][cc * 32], HDIM, b1.u);
      s0 = wmma_bf16(aq, b0, s0);
      s1 = wmma_bf16(aq, b1, s1);
    }

    // online softmax over the 32-key chunk (rows striped across 16 lanes)
    float mc[8];
    #pragma unroll
    for (int i = 0; i < 8; ++i) mc[i] = fmaxf(s0[i], s1[i]);
    #pragma unroll
    for (int off = 8; off >= 1; off >>= 1)
      #pragma unroll
      for (int i = 0; i < 8; ++i) mc[i] = fmaxf(mc[i], __shfl_xor(mc[i], off));

    float corr[8], rsum[8];
    #pragma unroll
    for (int i = 0; i < 8; ++i) {
      float nm = fmaxf(m_run[i], mc[i]);
      corr[i] = __expf(m_run[i] - nm);
      m_run[i] = nm;
      float p0 = __expf(s0[i] - nm);
      float p1 = __expf(s1[i] - nm);
      s0[i] = p0; s1[i] = p1;
      rsum[i] = p0 + p1;
    }
    #pragma unroll
    for (int off = 8; off >= 1; off >>= 1)
      #pragma unroll
      for (int i = 0; i < 8; ++i) rsum[i] += __shfl_xor(rsum[i], off);
    #pragma unroll
    for (int i = 0; i < 8; ++i) l_run[i] = l_run[i] * corr[i] + rsum[i];
    #pragma unroll
    for (int t = 0; t < 4; ++t)
      #pragma unroll
      for (int i = 0; i < 8; ++i) o[t][i] *= corr[i];

    // C-layout P -> A-layout via per-wave LDS bounce (wave-private: dscnt wait)
    #pragma unroll
    for (int i = 0; i < 8; ++i) {
      int r = i + hl * 8;
      Pl[wave][r][col]      = f2bf(s0[i]);
      Pl[wave][r][16 + col] = f2bf(s1[i]);
    }
    asm volatile("s_wait_dscnt 0x0" ::: "memory");
    BF16x16 pa; load_afrag(&Pl[wave][0][0], 32, pa.u);
    #pragma unroll
    for (int t = 0; t < 4; ++t) {
      BF16x16 bv; load_bfrag(&Vt[t * 16][0], 32, bv.u);
      o[t] = wmma_bf16(pa, bv, o[t]);
    }
  }

  #pragma unroll
  for (int i = 0; i < 8; ++i) {
    float inv = 1.0f / l_run[i];
    int row = blockIdx.x * 64 + wave * 16 + i + hl * 8;
    float* op = attn_out + (size_t)(b * SS + row) * DD + h * HDIM;
    #pragma unroll
    for (int t = 0; t < 4; ++t) op[t * 16 + col] = o[t][i] * inv;
  }
}

// ---------------------------------------------------------------------------
// LayerNorm of (a + r): one block per token row
// ---------------------------------------------------------------------------
__global__ __launch_bounds__(256) void ln_kernel(
    const float* __restrict__ a, const float* __restrict__ r,
    const float* __restrict__ g, const float* __restrict__ be,
    float* __restrict__ o)
{
  __shared__ float red[256];
  const int t = blockIdx.x, tid = threadIdx.x;
  const size_t base = (size_t)t * DD;
  float x0 = a[base + tid] + r[base + tid];
  float x1 = a[base + 256 + tid] + r[base + 256 + tid];
  red[tid] = x0 + x1; __syncthreads();
  for (int off = 128; off > 0; off >>= 1) { if (tid < off) red[tid] += red[tid + off]; __syncthreads(); }
  float mu = red[0] * (1.0f / DD);
  __syncthreads();
  float d0 = x0 - mu, d1 = x1 - mu;
  red[tid] = d0 * d0 + d1 * d1; __syncthreads();
  for (int off = 128; off > 0; off >>= 1) { if (tid < off) red[tid] += red[tid + off]; __syncthreads(); }
  float inv = rsqrtf(red[0] * (1.0f / DD) + 1e-5f);
  o[base + tid]       = d0 * inv * g[tid]       + be[tid];
  o[base + 256 + tid] = d1 * inv * g[256 + tid] + be[256 + tid];
}

// ---------------------------------------------------------------------------
// Router: logits, softmax/argmax, capacity scan, gather, scatter
// ---------------------------------------------------------------------------
__global__ __launch_bounds__(256) void router_logits_kernel(
    const float* __restrict__ h, const float* __restrict__ rw, float* __restrict__ logits)
{
  int idx = blockIdx.x * 256 + threadIdx.x;      // NT*E threads
  int t = idx >> 3, e = idx & 7;
  float acc = 0.0f;
  const float* hr = h + (size_t)t * DD;
  for (int d = 0; d < DD; ++d) acc += hr[d] * rw[d * EE + e];
  logits[idx] = acc;
}

__global__ __launch_bounds__(32) void router_softmax_kernel(
    const float* __restrict__ logits, float* __restrict__ probs,
    float* __restrict__ mask_out, int* __restrict__ amax)
{
  int t = blockIdx.x, lane = threadIdx.x;
  if (lane < EE) {
    float v = logits[t * EE + lane];
    float m = v;
    for (int off = 4; off >= 1; off >>= 1) m = fmaxf(m, __shfl_xor(m, off));
    float ex = __expf(v - m);
    float s = ex;
    for (int off = 4; off >= 1; off >>= 1) s += __shfl_xor(s, off);
    probs[t * EE + lane] = ex / s;
    mask_out[t * EE + lane] = 0.0f;
    float bv = v; int bi = lane;                   // first-max tiebreak
    for (int off = 1; off <= 4; off <<= 1) {
      float ov = __shfl_xor(bv, off); int oi = __shfl_xor(bi, off);
      if (ov > bv || (ov == bv && oi < bi)) { bv = ov; bi = oi; }
    }
    if (lane == 0) amax[t] = bi;
  }
}

// one wave per (expert, batch): in-order prefix scan with ballot/popc, capacity cap
__global__ __launch_bounds__(32) void router_scan_kernel(
    const int* __restrict__ amax, const float* __restrict__ probs,
    float* __restrict__ mask_out, int* __restrict__ sel,
    float* __restrict__ selp, int* __restrict__ counts)
{
  int be = blockIdx.x;                 // be = e*B + b
  int e = be >> 2, b = be & 3;
  int lane = threadIdx.x;
  int base = 0;
  for (int c = 0; c < SS / 32; ++c) {
    int t = c * 32 + lane;
    int gi = b * SS + t;
    bool flag = (amax[gi] == e);
    unsigned long long bal = __ballot(flag);
    int prefix = __popcll(bal & ((1ull << lane) - 1ull));
    if (flag) {
      int pos = base + prefix;
      if (pos < CAP) {
        float p = probs[gi * EE + e];
        mask_out[gi * EE + e] = p;
        sel[be * CAP + pos]  = t;
        selp[be * CAP + pos] = p;
      }
    }
    base += (int)__popcll(bal);
  }
  if (lane == 0) counts[be] = (base < CAP) ? base : CAP;
}

// gather routed tokens -> bf16 [E, B*CAP, D] (feeds async-staged FFN GEMM)
__global__ __launch_bounds__(256) void gather_kernel(
    const float* __restrict__ h, const int* __restrict__ sel,
    const int* __restrict__ counts, unsigned short* __restrict__ abuf)
{
  size_t idx = (size_t)blockIdx.x * 256 + threadIdx.x;   // E*B*CAP*D
  int d = (int)(idx & (DD - 1));
  int row = (int)(idx >> 9);
  int slot = row & (CAP - 1);
  int be = row >> 8;
  int b = be & 3;
  float v = 0.0f;
  if (slot < counts[be]) {
    int t = sel[row];
    v = h[(size_t)(b * SS + t) * DD + d];
  }
  abuf[idx] = f2bf(v);
}

__global__ __launch_bounds__(256) void zero_kernel(float* __restrict__ p) {
  p[(size_t)blockIdx.x * 256 + threadIdx.x] = 0.0f;
}

__global__ __launch_bounds__(256) void scatter_kernel(
    const float* __restrict__ y2, const int* __restrict__ sel,
    const float* __restrict__ selp, const int* __restrict__ counts,
    float* __restrict__ moe)
{
  size_t idx = (size_t)blockIdx.x * 256 + threadIdx.x;
  int d = (int)(idx & (DD - 1));
  int row = (int)(idx >> 9);
  int slot = row & (CAP - 1);
  int be = row >> 8;
  int b = be & 3;
  if (slot < counts[be]) {
    int t = sel[row];
    moe[(size_t)(b * SS + t) * DD + d] = selp[row] * y2[idx];   // top-1: exclusive
  }
}

// ---------------------------------------------------------------------------
extern "C" void kernel_launch(void* const* d_in, const int* in_sizes, int n_in,
                              void* d_out, int out_size, void* d_ws, size_t ws_size,
                              hipStream_t stream) {
  (void)in_sizes; (void)n_in; (void)out_size; (void)ws_size;
  const float* x     = (const float*)d_in[0];
  const float* in_w  = (const float*)d_in[1];
  const float* in_b  = (const float*)d_in[2];
  const float* out_w = (const float*)d_in[3];
  const float* out_b = (const float*)d_in[4];
  const float* ln1g  = (const float*)d_in[5];
  const float* ln1b  = (const float*)d_in[6];
  const float* ln2g  = (const float*)d_in[7];
  const float* ln2b  = (const float*)d_in[8];
  const float* rw    = (const float*)d_in[9];
  const float* wi    = (const float*)d_in[10];
  const float* wo    = (const float*)d_in[11];

  float* out        = (float*)d_out;                  // [B,S,D]
  float* logits_out = out + (size_t)NT * DD;          // [B,S,E]
  float* mask_out   = logits_out + (size_t)NT * EE;   // [B,S,E]

  // workspace layout (float units)
  float* ws = (float*)d_ws;
  constexpr size_t OFF_QKV   = 0;
  constexpr size_t OFF_AOUT  = OFF_QKV   + (size_t)NT * 3 * DD;
  constexpr size_t OFF_APROJ = OFF_AOUT  + (size_t)NT * DD;
  constexpr size_t OFF_H     = OFF_APROJ + (size_t)NT * DD;
  constexpr size_t OFF_PROBS = OFF_H     + (size_t)NT * DD;
  constexpr size_t OFF_ARGM  = OFF_PROBS + (size_t)NT * EE;
  constexpr size_t OFF_SEL   = OFF_ARGM  + (size_t)NT;
  constexpr size_t OFF_SELP  = OFF_SEL   + (size_t)EE * BB * CAP;
  constexpr size_t OFF_CNT   = OFF_SELP  + (size_t)EE * BB * CAP;
  constexpr size_t OFF_ABUF  = OFF_CNT   + 64;                              // bf16
  constexpr size_t OFF_ACT1  = OFF_ABUF  + (size_t)EE * BB * CAP * DD / 2;  // bf16
  constexpr size_t OFF_Y2    = OFF_ACT1  + (size_t)EE * BB * CAP * FF / 2;  // f32
  constexpr size_t OFF_MOE   = OFF_Y2    + (size_t)EE * BB * CAP * DD;
  // bf16 weight mirrors (ushort; sizes in float units = elems/2)
  constexpr size_t OFF_WIN   = OFF_MOE   + (size_t)NT * DD;
  constexpr size_t OFF_WOUT  = OFF_WIN   + (size_t)(3 * DD) * DD / 2;
  constexpr size_t OFF_WI    = OFF_WOUT  + (size_t)DD * DD / 2;
  constexpr size_t OFF_WO    = OFF_WI    + (size_t)EE * DD * FF / 2;

  float* qkv   = ws + OFF_QKV;
  float* aout  = ws + OFF_AOUT;
  float* aproj = ws + OFF_APROJ;
  float* hbuf  = ws + OFF_H;
  float* probs = ws + OFF_PROBS;
  int*   argm  = (int*)(ws + OFF_ARGM);
  int*   sel   = (int*)(ws + OFF_SEL);
  float* selp  = ws + OFF_SELP;
  int*   cnts  = (int*)(ws + OFF_CNT);
  unsigned short* abuf16 = (unsigned short*)(ws + OFF_ABUF);
  unsigned short* act116 = (unsigned short*)(ws + OFF_ACT1);
  float* y2    = ws + OFF_Y2;
  float* moe   = ws + OFF_MOE;
  unsigned short* win_bf  = (unsigned short*)(ws + OFF_WIN);
  unsigned short* wout_bf = (unsigned short*)(ws + OFF_WOUT);
  unsigned short* wiT_bf  = (unsigned short*)(ws + OFF_WI);   // [E][F][D]
  unsigned short* woT_bf  = (unsigned short*)(ws + OFF_WO);   // [E][D][F]

  // 0) one-shot weight conversion to bf16, FFN weights transposed to [N][K]
  cvt_bf16_kernel<<<(3 * DD * DD) / 256, 256, 0, stream>>>(in_w, win_bf, (long long)3 * DD * DD);
  cvt_bf16_kernel<<<(DD * DD) / 256, 256, 0, stream>>>(out_w, wout_bf, (long long)DD * DD);
  cvt_bf16_t_kernel<<<dim3((DD * FF) / 256, 1, EE), 256, 0, stream>>>(wi, wiT_bf, DD, FF); // [D][F]->[F][D]
  cvt_bf16_t_kernel<<<dim3((FF * DD) / 256, 1, EE), 256, 0, stream>>>(wo, woT_bf, FF, DD); // [F][D]->[D][F]

  // 1) qkv = x @ in_proj_w^T + b : M=4096 N=1536 K=512
  gemm_bf16_kernel<<<dim3(3 * DD / 64, NT / 128, 1), 256, 0, stream>>>(
      x, win_bf, in_b, qkv, DD, DD, DD, 3 * DD, 0, 0, 0, 0, 0, 0);
  // 2) flash attention
  attn_kernel<<<dim3(SS / 64, BB * HH), 128, 0, stream>>>(qkv, aout);
  // 3) out projection
  gemm_bf16_kernel<<<dim3(DD / 64, NT / 128, 1), 256, 0, stream>>>(
      aout, wout_bf, out_b, aproj, DD, DD, DD, DD, 0, 0, 0, 0, 0, 0);
  // 4) h = LN(x + attn)
  ln_kernel<<<NT, 256, 0, stream>>>(aproj, x, ln1g, ln1b, hbuf);
  // 5) router
  router_logits_kernel<<<NT * EE / 256, 256, 0, stream>>>(hbuf, rw, logits_out);
  router_softmax_kernel<<<NT, 32, 0, stream>>>(logits_out, probs, mask_out, argm);
  router_scan_kernel<<<EE * BB, 32, 0, stream>>>(argm, probs, mask_out, sel, selp, cnts);
  // 6) gather routed tokens -> bf16 [E, B*CAP, D]
  gather_kernel<<<(EE * BB * CAP * DD) / 256, 256, 0, stream>>>(hbuf, sel, cnts, abuf16);
  // 7) expert FFN, batched over experts; bf16 in / bf16 mid / f32 out
  gemm_bf16_kernel<<<dim3(FF / 64, (BB * CAP) / 128, EE), 256, 0, stream>>>(
      abuf16, wiT_bf, nullptr, act116, DD, DD, DD, FF, 1, 1, 1,
      (long long)BB * CAP * DD, (long long)DD * FF, (long long)BB * CAP * FF);
  gemm_bf16_kernel<<<dim3(DD / 64, (BB * CAP) / 128, EE), 256, 0, stream>>>(
      act116, woT_bf, nullptr, y2, FF, FF, FF, DD, 0, 1, 0,
      (long long)BB * CAP * FF, (long long)FF * DD, (long long)BB * CAP * DD);
  // 8) scatter prob-scaled outputs
  zero_kernel<<<(NT * DD) / 256, 256, 0, stream>>>(moe);
  scatter_kernel<<<(EE * BB * CAP * DD) / 256, 256, 0, stream>>>(y2, sel, selp, cnts, moe);
  // 9) out = LN(h + moe)
  ln_kernel<<<NT, 256, 0, stream>>>(moe, hbuf, ln2g, ln2b, out);
}
